// KANLayer_65644280152773
// MI455X (gfx1250) — compile-verified, matrix operand order checked by reading, not run on hardware
//
#include <hip/hip_runtime.h>

typedef __attribute__((ext_vector_type(16))) _Float16 v16h;
typedef __attribute__((ext_vector_type(8)))  _Float16 v8h;
typedef __attribute__((ext_vector_type(4)))  _Float16 v4h;
typedef __attribute__((ext_vector_type(8)))  float    v8f;

#define NB      8      // GRID_SIZE + SPLINE_ORDER = 5 + 3
#define ORDER   3
#define IN_DIM  64
#define OUT_DIM 64

// Cox-de Boor, order 3, knots = [-1,-1,-1,-1,-0.5,0,0.5,1,1,1,1,1].
// Fully unrolled; knot differences / reciprocals constant-fold, matching the
// reference's where(d != 0, (x - t)/d, 0) with division replaced by *(1/d)
// (1/d exact at compile time for these knot spacings except harmless rounding).
__device__ __forceinline__ void bspline8(float x, float B[NB]) {
  const float t[12] = {-1.f,-1.f,-1.f,-1.f,-0.5f,0.f,0.5f,1.f,1.f,1.f,1.f,1.f};
  float Bk[11];
#pragma unroll
  for (int i = 0; i < 11; ++i)
    Bk[i] = (x >= t[i] && x < t[i + 1]) ? 1.0f : 0.0f;
#pragma unroll
  for (int k = 1; k <= ORDER; ++k) {
#pragma unroll
    for (int i = 0; i < 11 - k; ++i) {
      const float d1 = t[i + k]     - t[i];
      const float d2 = t[i + k + 1] - t[i + 1];
      const float r1 = (d1 != 0.f) ? 1.0f / d1 : 0.0f;
      const float r2 = (d2 != 0.f) ? 1.0f / d2 : 0.0f;
      Bk[i] = (x - t[i]) * r1 * Bk[i] + (t[i + k + 1] - x) * r2 * Bk[i + 1];
    }
  }
#pragma unroll
  for (int i = 0; i < NB; ++i) B[i] = Bk[i];
}

__global__ __launch_bounds__(256)
void kan_wmma_kernel(const float* __restrict__ x,
                     const float* __restrict__ coeffs,
                     float* __restrict__ out) {
  // LDS: f16 copy of coeffs (64 KB) + per-wave x tiles (32 KB)
  __shared__ _Float16 sb[IN_DIM * OUT_DIM * NB];
  __shared__ float    sx[8 * 16 * IN_DIM];

  const int tid  = threadIdx.x;
  const int wave = tid >> 5;
  const int lane = tid & 31;
  const int row  = lane & 15;   // M-row within tile / N-col within n-tile
  const int half = lane >> 4;   // K-half selector per WMMA A/B VGPR layout

  const long mbase = ((long)blockIdx.x * 8 + wave) * 16;

  // ---- stage coeffs f32 -> f16 into LDS (whole block, coalesced) ----
  {
    const float4* c4  = (const float4*)coeffs;          // 8192 float4
    v4h*          sb4 = (v4h*)sb;
#pragma unroll
    for (int i = 0; i < 32; ++i) {
      const int idx = i * 256 + tid;
      const float4 v = c4[idx];
      v4h h;
      h[0] = (_Float16)v.x; h[1] = (_Float16)v.y;
      h[2] = (_Float16)v.z; h[3] = (_Float16)v.w;
      sb4[idx] = h;
    }
  }

  // ---- stage this wave's 16x64 x-tile into LDS (coalesced float4) ----
  {
    const float4* x4  = (const float4*)(x + mbase * IN_DIM);   // 256 float4
    float4*       sx4 = (float4*)(sx + wave * 16 * IN_DIM);
#pragma unroll
    for (int j = 0; j < 8; ++j) {
      const int fidx = j * 32 + lane;
      sx4[fidx] = x4[fidx];
    }
  }
  __syncthreads();

  const float* xw  = sx + wave * 16 * IN_DIM + row * IN_DIM;
  const v8h*   sb8 = (const v8h*)sb;   // sb8[i*OUT_DIM + o] = coeffs[i][o][0:8] in f16

  v8f acc[4] = {};   // 16x16 f32 C/D tiles for n-tiles 0..3

  for (int kc = 0; kc < 16; ++kc) {      // K chunks of 32 (= 4 input dims)
    const int d0 = kc * 4 + half;        // dims this lane owns (ISA A layout)
    const int d1 = d0 + 2;

    float Ba[NB], Bb[NB];
    bspline8(xw[d0], Ba);
    bspline8(xw[d1], Bb);

    v16h af;
#pragma unroll
    for (int i = 0; i < NB; ++i) {
      af[i]     = (_Float16)Ba[i];   // VGPR0-3: K = 8*half .. 8*half+7
      af[i + 8] = (_Float16)Bb[i];   // VGPR4-7: K = 16+8*half ..
    }

#pragma unroll
    for (int nt = 0; nt < 4; ++nt) {
      const int n = nt * 16 + row;   // output column this lane holds for B
      const v8h blo = sb8[d0 * OUT_DIM + n];   // ds_load_b128
      const v8h bhi = sb8[d1 * OUT_DIM + n];   // ds_load_b128
      v16h bf;
#pragma unroll
      for (int i = 0; i < 8; ++i) { bf[i] = blo[i]; bf[i + 8] = bhi[i]; }

      acc[nt] = __builtin_amdgcn_wmma_f32_16x16x32_f16(
          false, af, false, bf, (short)0, acc[nt], false, false);
    }
  }

  // ---- store: C/D layout -> VGPR r, lane L: row = mbase + half*8 + r, col = nt*16 + (L&15)
#pragma unroll
  for (int nt = 0; nt < 4; ++nt) {
#pragma unroll
    for (int r = 0; r < 8; ++r) {
      out[(mbase + half * 8 + r) * OUT_DIM + nt * 16 + row] = acc[nt][r];
    }
  }
}

extern "C" void kernel_launch(void* const* d_in, const int* in_sizes, int n_in,
                              void* d_out, int out_size, void* d_ws, size_t ws_size,
                              hipStream_t stream) {
  const float* x      = (const float*)d_in[0];
  const float* coeffs = (const float*)d_in[1];
  float*       out    = (float*)d_out;

  const int batch  = in_sizes[0] / IN_DIM;   // 131072
  const int blocks = batch / (8 * 16);       // 1024 blocks, 8 waves x 16 rows each

  hipLaunchKernelGGL(kan_wmma_kernel, dim3(blocks), dim3(256), 0, stream,
                     x, coeffs, out);
}
